// Network_79061757985000
// MI455X (gfx1250) — compile-verified, matrix operand order checked by reading, not run on hardware
//
#include <hip/hip_runtime.h>
#include <hip/hip_bf16.h>
#include <math.h>

typedef __attribute__((ext_vector_type(16))) _Float16 v16h;
typedef __attribute__((ext_vector_type(8)))  _Float16 v8h;
typedef __attribute__((ext_vector_type(8)))  float    v8f;

// Problem constants (match reference setup_inputs)
#define B_ROWS   65536
#define D_IN     784     // K1 (real)
#define K1_PAD   800     // padded to 25 * 32
#define D_H      300     // N1 (real)
#define N1_PAD   320     // padded to 20 * 16
#define D_OUT    10
#define N2_PAD   16
#define EPS      1e-5f

// Workspace layout (bytes)
#define OFF_W1B     0                      // _Float16 [N1_PAD][K1_PAD] = 512000 B
#define OFF_W2B     512000                 // _Float16 [N2_PAD][N1_PAD] = 10240 B
#define OFF_STATS1  522240                 // float [2*N1_PAD]  (sum, sumsq)
#define OFF_AB1     524800                 // float [2*N1_PAD]  (a, b)
#define OFF_STATS2  527360                 // float [2*N2_PAD]
#define OFF_H       528384                 // _Float16 [B_ROWS][N1_PAD] = 41943040 B
#define OFF_O       42471424               // float [B_ROWS][N2_PAD] = 4194304 B

// ---------------------------------------------------------------------------
// Kernel 1: binarize + pad weights into f16, zero the stats accumulators.
// ---------------------------------------------------------------------------
__global__ __launch_bounds__(256) void k_prep(const float* __restrict__ W1,
                                              const float* __restrict__ W2,
                                              _Float16* __restrict__ W1b,
                                              _Float16* __restrict__ W2b,
                                              float* __restrict__ stats1,
                                              float* __restrict__ stats2) {
  int idx = blockIdx.x * 256 + threadIdx.x;
  const int n1 = N1_PAD * K1_PAD;          // 256000
  const int n2 = N2_PAD * N1_PAD;          // 5120
  if (idx < n1) {
    int n = idx / K1_PAD, k = idx % K1_PAD;
    _Float16 v = (_Float16)0.0f;
    if (n < D_H && k < D_IN)
      v = (W1[n * D_IN + k] >= 0.0f) ? (_Float16)1.0f : (_Float16)-1.0f;
    W1b[idx] = v;
  } else if (idx < n1 + n2) {
    int q = idx - n1;
    int n = q / N1_PAD, k = q % N1_PAD;
    _Float16 v = (_Float16)0.0f;
    if (n < D_OUT && k < D_H)
      v = (W2[n * D_H + k] >= 0.0f) ? (_Float16)1.0f : (_Float16)-1.0f;
    W2b[q] = v;
  } else if (idx < n1 + n2 + 2 * N1_PAD) {
    stats1[idx - n1 - n2] = 0.0f;
  } else if (idx < n1 + n2 + 2 * N1_PAD + 2 * N2_PAD) {
    stats2[idx - n1 - n2 - 2 * N1_PAD] = 0.0f;
  }
}

// ---------------------------------------------------------------------------
// Kernel 2: GEMM1  h[B, 320] = x[B, 784] @ W1b^T   (f16 WMMA, f32 accum)
// Block: 256 thr = 8 waves. Block tile: 64 rows x 320 cols.
// Wave (wr,wc): wr=wave/2 row sub-tile (16 rows), wc=wave%2 col half (160 cols).
// W k-slab (320x32 f16) pipelined: regs -> LDS (fragment order), next slab's
// global loads issued before the compute phase so latency hides under WMMA.
// BN1 column sums/sumsq fused (exact f32 accumulators -> LDS -> global atomics).
// h stored as f16 (it is rounded to f16 by GEMM2's A-fragment anyway).
// ---------------------------------------------------------------------------
__global__ __launch_bounds__(256) void k_gemm1(const float* __restrict__ x,
                                               const _Float16* __restrict__ W1b,
                                               _Float16* __restrict__ h,
                                               float* __restrict__ stats1) {
  __shared__ _Float16 ldsW[20 * 512];      // 20 KB: 20 tiles * 32 lanes * 16 halfs
  __shared__ float    ldsS[2 * N1_PAD];    // block-level sum / sumsq

  const int tid  = threadIdx.x;
  const int wave = tid >> 5;
  const int lane = tid & 31;
  const int wr   = wave >> 1;              // 0..3
  const int wc   = wave & 1;               // 0..1
  const int hs   = lane >> 4;              // half-select: 0 or 1
  const int row0 = blockIdx.x * 64 + wr * 16;
  const int m    = row0 + (lane & 15);
  const float* xrow = x + (size_t)m * D_IN;

  for (int i = tid; i < 2 * N1_PAD; i += 256) ldsS[i] = 0.0f;

  // Per-thread W-slab chunk descriptors (5 x 16B chunks per thread).
  // LDS is written in exact B-fragment order:
  //   elem e<8  -> K = k0 + (lane/16)*8 + e ; elem e>=8 -> K = k0+16+(lane/16)*8+(e-8)
  //   lane n-index = tile*16 + lane%16 ; B[k][n] = W1b[n][k]
  const _Float16* gp[5];
  int lo[5];
#pragma unroll
  for (int it = 0; it < 5; ++it) {
    int cidx = tid + it * 256;
    int tile = cidx >> 6;
    int ln   = (cidx >> 1) & 31;
    int ch   = cidx & 1;
    int n    = tile * 16 + (ln & 15);
    gp[it] = W1b + n * K1_PAD + ch * 16 + (ln >> 4) * 8;
    lo[it] = tile * 512 + ln * 16 + ch * 8;
  }

  float4 r[5];
#pragma unroll
  for (int it = 0; it < 5; ++it) r[it] = *(const float4*)gp[it];   // slab 0

  v8f c[10];
#pragma unroll
  for (int t = 0; t < 10; ++t) c[t] = (v8f)0.0f;

  for (int step = 0; step < 25; ++step) {
    const int k0 = step * 32;

    __syncthreads();                       // previous slab fully consumed
#pragma unroll
    for (int it = 0; it < 5; ++it) *(float4*)&ldsW[lo[it]] = r[it];
    if (step + 1 < 25) {                   // issue next slab loads now; the
#pragma unroll                             // latency hides under the WMMAs
      for (int it = 0; it < 5; ++it)
        r[it] = *(const float4*)(gp[it] + (step + 1) * 32);
    }
    __syncthreads();

    // A fragment: 16x32 f16 from x (f32 -> f16 on the fly).
    v16h a;
    {
      const int kA = k0 + hs * 8;          // elements 0..7  (always < 784)
      const float4 u0 = *(const float4*)(xrow + kA);
      const float4 u1 = *(const float4*)(xrow + kA + 4);
      a[0] = (_Float16)u0.x; a[1] = (_Float16)u0.y;
      a[2] = (_Float16)u0.z; a[3] = (_Float16)u0.w;
      a[4] = (_Float16)u1.x; a[5] = (_Float16)u1.y;
      a[6] = (_Float16)u1.z; a[7] = (_Float16)u1.w;
      if (k0 + 16 < D_IN) {                // uniform branch: 784 ≡ 16 (mod 32)
        const int kB = k0 + 16 + hs * 8;   // elements 8..15
        const float4 v0 = *(const float4*)(xrow + kB);
        const float4 v1 = *(const float4*)(xrow + kB + 4);
        a[8]  = (_Float16)v0.x; a[9]  = (_Float16)v0.y;
        a[10] = (_Float16)v0.z; a[11] = (_Float16)v0.w;
        a[12] = (_Float16)v1.x; a[13] = (_Float16)v1.y;
        a[14] = (_Float16)v1.z; a[15] = (_Float16)v1.w;
      } else {
#pragma unroll
        for (int e = 8; e < 16; ++e) a[e] = (_Float16)0.0f;
      }
    }
    __builtin_prefetch(xrow + k0 + 32, 0, 3);   // next x slab, near-cache

#pragma unroll
    for (int t = 0; t < 10; ++t) {
      const int tile = wc * 10 + t;
      v16h b = *(const v16h*)&ldsW[tile * 512 + lane * 16];
      c[t] = __builtin_amdgcn_wmma_f32_16x16x32_f16(
          false, a, false, b, (short)0, c[t], false, false);
    }
  }

  // Fused BN1 statistics from exact f32 accumulators.
  // C/D layout: VGPR v -> M = v + 8*hs ; N = lane%16.
#pragma unroll
  for (int t = 0; t < 10; ++t) {
    float s = 0.0f, s2 = 0.0f;
#pragma unroll
    for (int v = 0; v < 8; ++v) { float val = c[t][v]; s += val; s2 += val * val; }
    s  += __shfl_xor(s, 16);               // combine M 0..7 with M 8..15
    s2 += __shfl_xor(s2, 16);
    if (lane < 16) {
      int col = wc * 160 + t * 16 + lane;
      atomicAdd(&ldsS[col], s);
      atomicAdd(&ldsS[N1_PAD + col], s2);
    }
  }

  // Store h as f16.
#pragma unroll
  for (int t = 0; t < 10; ++t) {
    const int nb = wc * 160 + t * 16 + (lane & 15);
#pragma unroll
    for (int v = 0; v < 8; ++v) {
      h[(size_t)(row0 + v + 8 * hs) * N1_PAD + nb] = (_Float16)c[t][v];
    }
  }

  __syncthreads();
  for (int i = tid; i < 2 * N1_PAD; i += 256) atomicAdd(&stats1[i], ldsS[i]);
}

// ---------------------------------------------------------------------------
// Kernel 3: fold BN1 into per-column affine a,b  (single block of 320)
// ---------------------------------------------------------------------------
__global__ __launch_bounds__(320) void k_final1(const float* __restrict__ stats1,
                                                const float* __restrict__ gamma1,
                                                const float* __restrict__ beta1,
                                                float* __restrict__ ab1) {
  const int j = threadIdx.x;
  const float inv = 1.0f / (float)B_ROWS;
  float mean = stats1[j] * inv;
  float var  = stats1[N1_PAD + j] * inv - mean * mean;
  float a = 0.0f, b = 0.0f;
  if (j < D_H) {
    a = gamma1[j] * rsqrtf(var + EPS);
    b = beta1[j] - mean * a;
  }
  ab1[j] = a;
  ab1[N1_PAD + j] = b;
}

// ---------------------------------------------------------------------------
// Kernel 4: GEMM2  o[B,16] = bn1(h)[B,320] @ W2b^T  (BN1 fused into A load,
// BN2 column stats fused out of the accumulators).
// Block: 256 thr = 8 waves; each wave: 16 rows x 16 cols, K = 320.
// ---------------------------------------------------------------------------
__global__ __launch_bounds__(256) void k_gemm2(const _Float16* __restrict__ h,
                                               const float* __restrict__ ab1,
                                               const _Float16* __restrict__ W2b,
                                               float* __restrict__ o,
                                               float* __restrict__ stats2) {
  __shared__ float ldsS2[2 * N2_PAD];

  const int tid  = threadIdx.x;
  const int wave = tid >> 5;
  const int lane = tid & 31;
  const int hs   = lane >> 4;
  const int row0 = blockIdx.x * 128 + wave * 16;
  const int m    = row0 + (lane & 15);
  const _Float16* hrow = h + (size_t)m * N1_PAD;
  const _Float16* wrow = W2b + (lane & 15) * N1_PAD;
  const float* a1 = ab1;
  const float* b1 = ab1 + N1_PAD;

  if (tid < 2 * N2_PAD) ldsS2[tid] = 0.0f;
  __syncthreads();

  v8f c = (v8f)0.0f;

#pragma unroll
  for (int step = 0; step < 10; ++step) {
    const int k0 = step * 32;
    const int kA = k0 + hs * 8;
    const int kB = k0 + 16 + hs * 8;

    v16h a;
    {
      const v8h hA = *(const v8h*)(hrow + kA);
      const v8h hB = *(const v8h*)(hrow + kB);
#pragma unroll
      for (int e = 0; e < 8; ++e) {
        a[e]     = (_Float16)((float)hA[e] * a1[kA + e] + b1[kA + e]);
        a[e + 8] = (_Float16)((float)hB[e] * a1[kB + e] + b1[kB + e]);
      }
    }

    v16h b;
    {
      const float4 w0 = *(const float4*)(wrow + kA);
      const float4 w1 = *(const float4*)(wrow + kB);
      *(float4*)&b       = w0;             // halfs 0..7  -> K kA..kA+7
      *((float4*)&b + 1) = w1;             // halfs 8..15 -> K kB..kB+7
    }

    c = __builtin_amdgcn_wmma_f32_16x16x32_f16(
        false, a, false, b, (short)0, c, false, false);
  }

  // Fused BN2 statistics.
  {
    float s = 0.0f, s2 = 0.0f;
#pragma unroll
    for (int v = 0; v < 8; ++v) { float val = c[v]; s += val; s2 += val * val; }
    s  += __shfl_xor(s, 16);
    s2 += __shfl_xor(s2, 16);
    if (lane < 16) {
      atomicAdd(&ldsS2[lane], s);
      atomicAdd(&ldsS2[N2_PAD + lane], s2);
    }
  }

#pragma unroll
  for (int v = 0; v < 8; ++v) {
    o[(size_t)(row0 + v + 8 * hs) * N2_PAD + (lane & 15)] = c[v];
  }

  __syncthreads();
  if (tid < 2 * N2_PAD) atomicAdd(&stats2[tid], ldsS2[tid]);
}

// ---------------------------------------------------------------------------
// Kernel 5: apply BN2, write final output [65536 x 10] f32
// ---------------------------------------------------------------------------
__global__ __launch_bounds__(256) void k_apply2(const float* __restrict__ o,
                                                const float* __restrict__ stats2,
                                                const float* __restrict__ gamma2,
                                                const float* __restrict__ beta2,
                                                float* __restrict__ out) {
  const int idx = blockIdx.x * 256 + threadIdx.x;   // < 655360
  const int j = idx % D_OUT;
  const int r = idx / D_OUT;
  const float inv = 1.0f / (float)B_ROWS;
  const float mean = stats2[j] * inv;
  const float var  = stats2[N2_PAD + j] * inv - mean * mean;
  const float sc   = gamma2[j] * rsqrtf(var + EPS);
  out[idx] = (o[(size_t)r * N2_PAD + j] - mean) * sc + beta2[j];
}

// ---------------------------------------------------------------------------
extern "C" void kernel_launch(void* const* d_in, const int* in_sizes, int n_in,
                              void* d_out, int out_size, void* d_ws, size_t ws_size,
                              hipStream_t stream) {
  const float* x      = (const float*)d_in[0];
  const float* W1     = (const float*)d_in[1];
  const float* gamma1 = (const float*)d_in[2];
  const float* beta1  = (const float*)d_in[3];
  const float* W2     = (const float*)d_in[4];
  const float* gamma2 = (const float*)d_in[5];
  const float* beta2  = (const float*)d_in[6];
  float* out = (float*)d_out;

  char* ws = (char*)d_ws;
  _Float16* W1b   = (_Float16*)(ws + OFF_W1B);
  _Float16* W2b   = (_Float16*)(ws + OFF_W2B);
  float* stats1   = (float*)(ws + OFF_STATS1);
  float* ab1      = (float*)(ws + OFF_AB1);
  float* stats2   = (float*)(ws + OFF_STATS2);
  _Float16* h     = (_Float16*)(ws + OFF_H);
  float* o        = (float*)(ws + OFF_O);

  // 1) binarize/pad weights to f16 + zero stats
  {
    const int total = N1_PAD * K1_PAD + N2_PAD * N1_PAD + 2 * N1_PAD + 2 * N2_PAD;
    k_prep<<<(total + 255) / 256, 256, 0, stream>>>(W1, W2, W1b, W2b, stats1, stats2);
  }
  // 2) GEMM1 (WMMA f16) + fused BN1 stats: 65536/64 = 1024 blocks
  k_gemm1<<<B_ROWS / 64, 256, 0, stream>>>(x, W1b, h, stats1);
  // 3) BN1 fold
  k_final1<<<1, 320, 0, stream>>>(stats1, gamma1, beta1, ab1);
  // 4) GEMM2 (WMMA f16, BN1 fused in, BN2 stats fused out): 512 blocks
  k_gemm2<<<B_ROWS / 128, 256, 0, stream>>>(h, ab1, W2b, o, stats2);
  // 5) BN2 apply -> d_out
  k_apply2<<<(B_ROWS * D_OUT) / 256, 256, 0, stream>>>(o, stats2, gamma2, beta2, out);
}
